// AttnA_67216238182978
// MI455X (gfx1250) — matrix operand order repaired and checked by reading, not compiled, not run on hardware
//
#include <hip/hip_runtime.h>

// MI455X / gfx1250, wave32. D = A(16x32 bf16) * B(32x16 bf16) + C(16x16 f32)
typedef __attribute__((ext_vector_type(16))) __bf16 v16bf;
typedef __attribute__((ext_vector_type(8)))  __bf16 v8bf;
typedef __attribute__((ext_vector_type(8)))  float  v8f;

#define T_SEQ 2048
#define DIMC  512
#define NHQ   8
#define NKVH  4
#define HDIM  64

__device__ __forceinline__ v8f wmma_bf16(v16bf a, v16bf b, v8f c) {
  // (neg_a, A, neg_b, B, c_mod, C, reuse_a, reuse_b)
  return __builtin_amdgcn_wmma_f32_16x16x32_bf16(false, a, false, b, (short)0, c,
                                                 false, false);
}

// CDNA5 async copy: global -> LDS, 16 bytes per active lane.
// lds_off = raw LDS byte address (low 32 bits of the generic pointer).
__device__ __forceinline__ void async_load_b128(unsigned lds_off, const void* gaddr) {
  asm volatile("global_load_async_to_lds_b128 %0, %1, off"
               :: "v"(lds_off), "v"(gaddr) : "memory");
}
__device__ __forceinline__ void wait_async0() {
  asm volatile("s_wait_asynccnt 0x0" ::: "memory");
}

// ---------------------------------------------------------------------------
// Kernel 0: one-shot fp32 -> bf16 pack (x, [Wq|Wk|Wv], Wp).
// ---------------------------------------------------------------------------
__global__ __launch_bounds__(256) void k_cvt(const float* __restrict__ src,
                                             __bf16* __restrict__ dst) {
  const size_t i = ((size_t)blockIdx.x * 256 + threadIdx.x) * 8;
  const float4 a = *(const float4*)(src + i);
  const float4 b = *(const float4*)(src + i + 4);
  v8bf o;
  o[0] = (__bf16)a.x; o[1] = (__bf16)a.y; o[2] = (__bf16)a.z; o[3] = (__bf16)a.w;
  o[4] = (__bf16)b.x; o[5] = (__bf16)b.y; o[6] = (__bf16)b.z; o[7] = (__bf16)b.w;
  *(v8bf*)(dst + i) = o;
}

// ---------------------------------------------------------------------------
// Kernel 1: QKV projection (bf16 WMMA) fused with RMSNorm + RoPE.
// Block = 128 threads = 4 waves; wave w owns rows [rowbase, rowbase+16),
// block owns cols [colbase, colbase+64). The 64x32 bf16 weight tile is staged
// once per k-step through LDS with global_load_async_to_lds_b128 and shared
// by all 4 waves.
// ---------------------------------------------------------------------------
__global__ __launch_bounds__(128) void k_qkv(const __bf16* __restrict__ Xr,
                                             const __bf16* __restrict__ Wb,  // [1024][512]
                                             __bf16* __restrict__ Qr,
                                             __bf16* __restrict__ Kr,
                                             __bf16* __restrict__ Vt) {
  __shared__ __bf16 wtile[64][32];  // 64 cols x 32 K = 4 KB
  const int tid  = threadIdx.x;
  const int lane = tid & 31;
  const int wave = tid >> 5;
  const int hlf  = lane >> 4;   // which 16-lane half
  const int nid  = lane & 15;   // position within half
  const int rowbase = blockIdx.x * 64 + wave * 16;
  const int colbase = blockIdx.y * 64;

  v8f acc[4] = {};

  // per-thread async-copy mapping: bytes [tid*16) and [tid*16+2048) of the tile
  const unsigned off0 = (unsigned)tid * 16u;
  const unsigned off1 = off0 + 2048u;
  const unsigned ldsbase = (unsigned)(unsigned long long)&wtile[0][0];
  const __bf16* g0row = Wb + (size_t)(colbase + (off0 >> 6)) * DIMC + ((off0 & 63u) >> 1);
  const __bf16* g1row = Wb + (size_t)(colbase + (off1 >> 6)) * DIMC + ((off1 & 63u) >> 1);

  const __bf16* xrow = Xr + (size_t)(rowbase + nid) * DIMC;
  for (int kk = 0; kk < DIMC; kk += 32) {
    // stage B tile (64 cols x 32 K) into LDS via async copy
    async_load_b128(ldsbase + off0, g0row + kk);
    async_load_b128(ldsbase + off1, g1row + kk);
    wait_async0();
    __syncthreads();

    // A fragment: lane row = rowbase+nid, K(e) = 16*hlf + e
    v16bf a;
    const __bf16* xp = xrow + kk + 16 * hlf;
    #pragma unroll
    for (int e = 0; e < 16; ++e) a[e] = xp[e];
    #pragma unroll
    for (int f = 0; f < 4; ++f) {
      // B fragment: lane col = 16f+nid, K = 16*hlf + e (contiguous in LDS row)
      const __bf16* wp = &wtile[16 * f + nid][16 * hlf];
      v16bf b;
      #pragma unroll
      for (int e = 0; e < 16; ++e) b[e] = wp[e];
      acc[f] = wmma_bf16(a, b, acc[f]);
    }
    __syncthreads();  // protect wtile before next k-step's async writes
  }

  // RMSNorm + RoPE for q/k tiles (one wave tile == one full head's 64 dims)
  if (colbase < DIMC + 256) {
    const float inv0 = __powf(10000.0f, -(float)nid        * (1.0f / 32.0f));
    const float inv1 = __powf(10000.0f, -(float)(nid + 16) * (1.0f / 32.0f));
    #pragma unroll
    for (int r = 0; r < 8; ++r) {
      float ss = 0.0f;
      #pragma unroll
      for (int f = 0; f < 4; ++f) ss += acc[f][r] * acc[f][r];
      ss += __shfl_xor(ss, 1, 16);
      ss += __shfl_xor(ss, 2, 16);
      ss += __shfl_xor(ss, 4, 16);
      ss += __shfl_xor(ss, 8, 16);
      const float sc = rsqrtf(ss * (1.0f / 64.0f) + 1.1920928955078125e-07f);
      #pragma unroll
      for (int f = 0; f < 4; ++f) acc[f][r] *= sc;
      // RoPE: pair (hd, hd+32); hd = 16*f + nid for f in {0,1}
      const float t = (float)((rowbase + r + 8 * hlf) & (T_SEQ - 1));
      const float a0 = t * inv0, a1 = t * inv1;
      const float c0 = __cosf(a0), s0 = __sinf(a0);
      const float c1 = __cosf(a1), s1 = __sinf(a1);
      float x1 = acc[0][r], x2 = acc[2][r];
      acc[0][r] =  x1 * c0 + x2 * s0;
      acc[2][r] = -x1 * s0 + x2 * c0;
      x1 = acc[1][r]; x2 = acc[3][r];
      acc[1][r] =  x1 * c1 + x2 * s1;
      acc[3][r] = -x1 * s1 + x2 * c1;
    }
  }

  const int b  = rowbase >> 11;          // / T_SEQ
  const int t0 = rowbase & (T_SEQ - 1);
  if (colbase < DIMC) {                  // Q: [B, NH, T, HD]
    const int h = colbase >> 6;
    __bf16* qb = Qr + (size_t)(b * NHQ + h) * T_SEQ * HDIM;
    #pragma unroll
    for (int r = 0; r < 8; ++r) {
      const int t = t0 + r + 8 * hlf;
      #pragma unroll
      for (int f = 0; f < 4; ++f)
        qb[(size_t)t * HDIM + 16 * f + nid] = (__bf16)acc[f][r];
    }
  } else if (colbase < DIMC + 256) {     // K: [B, NKV, T, HD]
    const int h = (colbase - DIMC) >> 6;
    __bf16* kp = Kr + (size_t)(b * NKVH + h) * T_SEQ * HDIM;
    #pragma unroll
    for (int r = 0; r < 8; ++r) {
      const int t = t0 + r + 8 * hlf;
      #pragma unroll
      for (int f = 0; f < 4; ++f)
        kp[(size_t)t * HDIM + 16 * f + nid] = (__bf16)acc[f][r];
    }
  } else {                               // V^T: [B, NKV, HD, T]
    const int h = (colbase - DIMC - 256) >> 6;
    __bf16* vp = Vt + (size_t)(b * NKVH + h) * HDIM * T_SEQ;
    #pragma unroll
    for (int r = 0; r < 8; ++r) {
      const int t = t0 + r + 8 * hlf;
      #pragma unroll
      for (int f = 0; f < 4; ++f)
        vp[(size_t)(16 * f + nid) * T_SEQ + t] = (__bf16)acc[f][r];
    }
  }
}

// ---------------------------------------------------------------------------
// Kernel 2: causal flash attention. Block = 4 waves; each wave owns 16 query
// rows of one (b, h). 32 keys per iteration: 4 WMMAs for S, 4 for P*V^T.
// ---------------------------------------------------------------------------
__global__ __launch_bounds__(128) void k_attn(const __bf16* __restrict__ Qr,
                                              const __bf16* __restrict__ Kr,
                                              const __bf16* __restrict__ Vt,
                                              __bf16* __restrict__ Yw) {
  __shared__ __bf16 plds[4][16][32];  // per-wave P tile (C-layout -> A-layout)
  const int lane = threadIdx.x & 31;
  const int wave = threadIdx.x >> 5;
  const int hlf  = lane >> 4;
  const int nid  = lane & 15;

  const int qt = blockIdx.x & 31;      // T / 64 = 32 query tiles
  const int bh = blockIdx.x >> 5;
  const int b  = bh >> 3;
  const int h  = bh & 7;
  const int hk = h >> 1;               // GQA: rep = NH/NKV = 2
  const int q0 = qt * 64 + wave * 16;

  const __bf16* Qp = Qr + (size_t)(b * NHQ + h)  * T_SEQ * HDIM;
  const __bf16* Kp = Kr + (size_t)(b * NKVH + hk) * T_SEQ * HDIM;
  const __bf16* Vp = Vt + (size_t)(b * NKVH + hk) * HDIM * T_SEQ;

  // Q A-fragments (K-dim = HD = 64 -> two k-steps of 32), held in VGPRs
  v16bf qa[2];
  {
    const __bf16* qrow = Qp + (size_t)(q0 + nid) * HDIM;
    #pragma unroll
    for (int ks = 0; ks < 2; ++ks) {
      const __bf16* qp = qrow + ks * 32 + 16 * hlf;
      #pragma unroll
      for (int e = 0; e < 16; ++e) qa[ks][e] = qp[e];
    }
  }

  v8f o[4] = {};
  float mrow[8], lrow[8];
  #pragma unroll
  for (int r = 0; r < 8; ++r) { mrow[r] = -3.0e38f; lrow[r] = 0.0f; }

  for (int kb = 0; kb < q0 + 16; kb += 32) {
    // S = Q * K^T : B-fragment lane col = key, K-dim contiguous along hd
    v8f s[2] = {};
    #pragma unroll
    for (int tl = 0; tl < 2; ++tl) {
      #pragma unroll
      for (int ks = 0; ks < 2; ++ks) {
        const __bf16* kp = Kp + (size_t)(kb + 16 * tl + nid) * HDIM + ks * 32 + 16 * hlf;
        v16bf kf;
        #pragma unroll
        for (int e = 0; e < 16; ++e) kf[e] = kp[e];
        s[tl] = wmma_bf16(qa[ks], kf, s[tl]);
      }
    }
    // online softmax over 32 keys (row m = r + 8*hlf, col = key)
    #pragma unroll
    for (int r = 0; r < 8; ++r) {
      const int q = q0 + r + 8 * hlf;
      float sm = -3.0e38f;
      #pragma unroll
      for (int tl = 0; tl < 2; ++tl) {
        const int key = kb + 16 * tl + nid;
        float sv = s[tl][r] * 0.125f;        // 1/sqrt(HD)
        if (key > q) sv = -3.0e38f;          // causal mask
        s[tl][r] = sv;
        sm = fmaxf(sm, sv);
      }
      sm = fmaxf(sm, __shfl_xor(sm, 1, 16));
      sm = fmaxf(sm, __shfl_xor(sm, 2, 16));
      sm = fmaxf(sm, __shfl_xor(sm, 4, 16));
      sm = fmaxf(sm, __shfl_xor(sm, 8, 16));
      const float mn = fmaxf(mrow[r], sm);
      const float al = __expf(mrow[r] - mn);
      mrow[r] = mn;
      float ps = 0.0f;
      #pragma unroll
      for (int tl = 0; tl < 2; ++tl) {
        const float p = __expf(s[tl][r] - mn);
        s[tl][r] = p;
        ps += p;
      }
      ps += __shfl_xor(ps, 1, 16);
      ps += __shfl_xor(ps, 2, 16);
      ps += __shfl_xor(ps, 4, 16);
      ps += __shfl_xor(ps, 8, 16);
      lrow[r] = lrow[r] * al + ps;
      #pragma unroll
      for (int f = 0; f < 4; ++f) o[f][r] *= al;
      const int m = r + 8 * hlf;
      plds[wave][m][nid]      = (__bf16)s[0][r];
      plds[wave][m][16 + nid] = (__bf16)s[1][r];
    }
    asm volatile("s_wait_dscnt 0x0" ::: "memory");  // LDS write -> read (cross-lane)
    // Re-read P in A-fragment layout (16x32, K = 32 keys)
    v16bf pa;
    {
      const __bf16* pp = &plds[wave][nid][16 * hlf];
      #pragma unroll
      for (int e = 0; e < 16; ++e) pa[e] = pp[e];
    }
    // O += P * V : B-fragment from V^T, contiguous along key dim per lane
    #pragma unroll
    for (int f = 0; f < 4; ++f) {
      const __bf16* vp = Vp + (size_t)(16 * f + nid) * T_SEQ + kb + 16 * hlf;
      v16bf vf;
      #pragma unroll
      for (int e = 0; e < 16; ++e) vf[e] = vp[e];
      o[f] = wmma_bf16(pa, vf, o[f]);
    }
  }

  // Normalize and store y[b, t, h*64 + hd] as bf16
  #pragma unroll
  for (int r = 0; r < 8; ++r) {
    const float inv = 1.0f / lrow[r];
    const int q = q0 + r + 8 * hlf;
    __bf16* yp = Yw + (size_t)(b * T_SEQ + q) * DIMC + h * HDIM;
    #pragma unroll
    for (int f = 0; f < 4; ++f)
      yp[16 * f + nid] = (__bf16)(o[f][r] * inv);
  }
}

// ---------------------------------------------------------------------------
// Kernel 3: output projection out = Y * Wp^T (bf16 A/B, f32 out)
// ---------------------------------------------------------------------------
__global__ __launch_bounds__(128) void k_proj(const __bf16* __restrict__ Yw,
                                              const __bf16* __restrict__ Wpb,
                                              float* __restrict__ out) {
  const int lane = threadIdx.x & 31;
  const int wave = threadIdx.x >> 5;
  const int hlf  = lane >> 4;
  const int nid  = lane & 15;
  const int rowbase = blockIdx.x * 64 + wave * 16;
  const int colbase = blockIdx.y * 64;

  v8f acc[4] = {};
  const __bf16* yrow = Yw + (size_t)(rowbase + nid) * DIMC;
  for (int kk = 0; kk < DIMC; kk += 32) {
    v16bf a;
    const __bf16* yp = yrow + kk + 16 * hlf;
    #pragma unroll
    for (int e = 0; e < 16; ++e) a[e] = yp[e];
    #pragma unroll
    for (int f = 0; f < 4; ++f) {
      const __bf16* wp = Wpb + (size_t)(colbase + 16 * f + nid) * DIMC + kk + 16 * hlf;
      __builtin_prefetch(wp + 32, 0, 0);
      v16bf b;
      #pragma unroll
      for (int e = 0; e < 16; ++e) b[e] = wp[e];
      acc[f] = wmma_bf16(a, b, acc[f]);
    }
  }
  #pragma unroll
  for (int r = 0; r < 8; ++r) {
    float* op = out + (size_t)(rowbase + r + 8 * hlf) * DIMC + colbase;
    #pragma unroll
    for (int f = 0; f < 4; ++f) op[16 * f + nid] = acc[f][r];
  }
}

// ---------------------------------------------------------------------------
extern "C" void kernel_launch(void* const* d_in, const int* in_sizes, int n_in,
                              void* d_out, int out_size, void* d_ws, size_t ws_size,
                              hipStream_t stream) {
  const float* x  = (const float*)d_in[0];
  const float* Wq = (const float*)d_in[1];
  const float* Wk = (const float*)d_in[2];
  const float* Wv = (const float*)d_in[3];
  const float* Wp = (const float*)d_in[4];
  float* out = (float*)d_out;

  char* ws = (char*)d_ws;
  const size_t MB = 1024 * 1024;
  __bf16* Qr  = (__bf16*)(ws);             // 16 MB  [B,NH,T,HD]
  __bf16* Kr  = (__bf16*)(ws + 16 * MB);   //  8 MB  [B,NKV,T,HD]
  __bf16* Vt  = (__bf16*)(ws + 24 * MB);   //  8 MB  [B,NKV,HD,T]
  __bf16* Yw  = (__bf16*)(ws + 32 * MB);   // 16 MB  [B,T,DIM]
  __bf16* Xr  = (__bf16*)(ws + 48 * MB);   // 16 MB  x in bf16
  __bf16* Wb  = (__bf16*)(ws + 64 * MB);   //  1 MB  [Wq|Wk|Wv] 1024x512 bf16
  __bf16* Wpb = (__bf16*)(ws + 66 * MB);   // 0.5 MB Wp 512x512 bf16

  const dim3 blk(128);
  // one-shot bf16 packs (sizes all divisible by 2048 elements)
  k_cvt<<<dim3(4096), 256, 0, stream>>>(x,  Xr);             // 8*2048*512
  k_cvt<<<dim3(128),  256, 0, stream>>>(Wq, Wb);             // 512*512
  k_cvt<<<dim3(64),   256, 0, stream>>>(Wk, Wb + 512 * 512); // 256*512
  k_cvt<<<dim3(64),   256, 0, stream>>>(Wv, Wb + 768 * 512); // 256*512
  k_cvt<<<dim3(128),  256, 0, stream>>>(Wp, Wpb);            // 512*512

  // rows = B*T = 16384 (64/block); cols = 512+256+256 = 1024 (64/block)
  k_qkv<<<dim3(256, 16), blk, 0, stream>>>(Xr, Wb, Qr, Kr, Vt);
  // B*NH = 64 (b,h) pairs x 32 query tiles of 64
  k_attn<<<dim3(2048), blk, 0, stream>>>(Qr, Kr, Vt, Yw);
  // rows = 16384, cols = 512
  k_proj<<<dim3(256, 8), blk, 0, stream>>>(Yw, Wpb, out);
}
